// Net_54924041781297
// MI455X (gfx1250) — compile-verified, matrix operand order checked by reading, not run on hardware
//
#include <hip/hip_runtime.h>
#include <hip/hip_bf16.h>
#include <math.h>

// ---------------------------------------------------------------------------
// CDNA5 (gfx1250) LieConv classifier pipeline.
//  * Heavy GEMMs -> v_wmma_f32_16x16x32_bf16, f32 accumulate, 64x128 tiles
//    (4 waves x (2x4) 16x16 tiles = 8 WMMA per k-step per wave).
//  * Weight-stationary bf16: weights pre-converted (transposed) once; B tiles
//    DMA'd into LDS by the Tensor Data Mover (tensor_load_to_lds + TENSORcnt).
//  * A tiles staged fp32->bf16 with unguarded float4 loads on interior tiles;
//    edge tiles use clamp+select (no exec-mask branching).
//  * FPS / kNN / gathers / BatchNorm remain VALU/LDS kernels.
// ---------------------------------------------------------------------------

typedef __attribute__((ext_vector_type(16))) __bf16 v16bf;
typedef __attribute__((ext_vector_type(8)))  float  v8f;
typedef __attribute__((ext_vector_type(4)))  unsigned int u32x4;
typedef __attribute__((ext_vector_type(8)))  int i32x8;
typedef __attribute__((ext_vector_type(4)))  int i32x4;

#if __has_builtin(__builtin_amdgcn_tensor_load_to_lds)
#define HAVE_TDM 1
#else
#define HAVE_TDM 0
#endif

__device__ __forceinline__ unsigned short f2bf(float f) {
    unsigned int u = __float_as_uint(f);
    unsigned int r = u + 0x7FFFu + ((u >> 16) & 1u);   // round-to-nearest-even
    return (unsigned short)(r >> 16);
}

#define TM 64
#define TN 128
#define TK 32

// ---- A tile staging: fp32 [M,K](lda) -> bf16 LDS [64 x 32] ----------------
__device__ __forceinline__ void stage_A(const float* __restrict__ A, int lda, int M, int K,
                                        int rowBase, int k0, int kTiles, int kb,
                                        unsigned short* As, int tid)
{
    const int r     = tid >> 1;
    const int cBase = (tid & 1) * 16;
    const int grow  = rowBase + r;
    if (grow < M && (k0 + TK) <= K && ((lda & 3) == 0)) {
        const float* Arow = A + (size_t)grow * lda;
        if ((kb + 1) < kTiles)
            __builtin_prefetch(Arow + k0 + TK, 0, 1);          // global_prefetch_b8
        #pragma unroll
        for (int c = 0; c < 16; c += 4) {
            const float4 v = *(const float4*)(Arow + k0 + cBase + c);
            const unsigned lo = ((unsigned)f2bf(v.y) << 16) | (unsigned)f2bf(v.x);
            const unsigned hi = ((unsigned)f2bf(v.w) << 16) | (unsigned)f2bf(v.z);
            *(unsigned*)&As[r * TK + cBase + c]     = lo;
            *(unsigned*)&As[r * TK + cBase + c + 2] = hi;
        }
    } else {
        // edge tile: clamp addresses, select zero -- straight-line codegen
        const int rowC = grow < M ? grow : (M - 1);
        const float* Arow = A + (size_t)rowC * lda;
        #pragma unroll
        for (int c = 0; c < 16; ++c) {
            const int kk  = k0 + cBase + c;
            const int kkC = kk < K ? kk : (K - 1);
            float v = Arow[kkC];
            v = (grow < M && kk < K) ? v : 0.0f;
            As[r * TK + cBase + c] = f2bf(v);
        }
    }
}

// ---- B tile staging from fp32 W[K,N]: coalesced rows, transposed store ----
__device__ __forceinline__ void stage_B_f32(const float* __restrict__ W, int N, int K,
                                            int colBase, int k0, unsigned short* Bs, int tid)
{
    const int kr    = tid >> 2;          // 0..31 : k within tile
    const int nBase = (tid & 3) * 32;    // 0/32/64/96
    const int kk    = k0 + kr;
    if (kk < K && (colBase + TN) <= N && ((N & 3) == 0)) {
        const float* Wrow = W + (size_t)kk * N + colBase + nBase;
        #pragma unroll
        for (int j = 0; j < 32; j += 4) {
            const float4 v = *(const float4*)(Wrow + j);
            Bs[(nBase + j + 0) * TK + kr] = f2bf(v.x);
            Bs[(nBase + j + 1) * TK + kr] = f2bf(v.y);
            Bs[(nBase + j + 2) * TK + kr] = f2bf(v.z);
            Bs[(nBase + j + 3) * TK + kr] = f2bf(v.w);
        }
    } else {
        const int kkC = kk < K ? kk : (K - 1);
        const float* Wrow = W + (size_t)kkC * N;
        #pragma unroll
        for (int j = 0; j < 32; ++j) {
            const int n  = colBase + nBase + j;
            const int nC = n < N ? n : (N - 1);
            float v = Wrow[nC];
            v = (kk < K && n < N) ? v : 0.0f;
            Bs[(nBase + j) * TK + kr] = f2bf(v);
        }
    }
}

// ---- WMMA compute + epilogue (shared by both GEMM kernels) ----------------
__device__ __forceinline__ void wmma_tile(const unsigned short* As, const unsigned short* Bs,
                                          int lane, int wm, int wn, v8f acc[2][4])
{
    const int khalf = (lane >> 4) * 16;  // this code's k order: lane>=16 -> K=16..31 (contiguous)
    const int rsub  = lane & 15;
    v16bf bfrag[4];
    #pragma unroll
    for (int ni = 0; ni < 4; ++ni)
        bfrag[ni] = *(const v16bf*)&Bs[(wn + ni * 16 + rsub) * TK + khalf];
    #pragma unroll
    for (int mi = 0; mi < 2; ++mi) {
        const v16bf afrag = *(const v16bf*)&As[(wm + mi * 16 + rsub) * TK + khalf];
        #pragma unroll
        for (int ni = 0; ni < 4; ++ni)
            acc[mi][ni] = __builtin_amdgcn_wmma_f32_16x16x32_bf16(
                false, afrag, false, bfrag[ni], (short)0, acc[mi][ni], false, false);
    }
}

__device__ __forceinline__ void epilogue(v8f acc[2][4], const float* bias, float* C, int ldc,
                                         int M, int N, int rowBase, int colBase,
                                         int lane, int wm, int wn, int relu)
{
    const int nsub  = lane & 15;         // C/D layout: lane&15 = N, lane>=16 -> M+8
    const int mhalf = (lane >> 4) * 8;
    #pragma unroll
    for (int mi = 0; mi < 2; ++mi)
        #pragma unroll
        for (int ni = 0; ni < 4; ++ni)
            #pragma unroll
            for (int r = 0; r < 8; ++r) {
                const int m = rowBase + wm + mi * 16 + mhalf + r;
                const int n = colBase + wn + ni * 16 + nsub;
                if (m < M && n < N) {
                    float v = acc[mi][ni][r];
                    if (bias) v += bias[n];
                    if (relu) v = fmaxf(v, 0.0f);
                    C[(size_t)m * ldc + n] = v;
                }
            }
}

// ---- generic GEMM: C = act(A@W + b), fp32 W --------------------------------
__global__ __launch_bounds__(128)
void gemm_bias_act(const float* __restrict__ A, int lda,
                   const float* __restrict__ W,
                   const float* __restrict__ bias,
                   float* __restrict__ C, int ldc,
                   int M, int N, int K, int relu)
{
    __shared__ __align__(32) unsigned short As[TM * TK];
    __shared__ __align__(32) unsigned short Bs[TN * TK];
    const int tid = threadIdx.x, lane = tid & 31, wave = tid >> 5;
    const int wm = (wave >> 1) * 32, wn = (wave & 1) * 64;   // wave: 32x64 region
    const int rowBase = blockIdx.y * TM, colBase = blockIdx.x * TN;
    v8f acc[2][4] = {};
    const int kTiles = (K + TK - 1) / TK;
    for (int kb = 0; kb < kTiles; ++kb) {
        const int k0 = kb * TK;
        stage_A(A, lda, M, K, rowBase, k0, kTiles, kb, As, tid);
        stage_B_f32(W, N, K, colBase, k0, Bs, tid);
        __syncthreads();
        wmma_tile(As, Bs, lane, wm, wn, acc);
        __syncthreads();
    }
    epilogue(acc, bias, C, ldc, M, N, rowBase, colBase, lane, wm, wn, relu);
}

// ---- TDM GEMM: bf16 weight-stationary, B tiles DMA'd to LDS ----------------
// Wt is [N,K] bf16 (transposed weight). Requires K%32==0, N%128==0.
__global__ __launch_bounds__(128)
void gemm_bias_act_tdm(const float* __restrict__ A, int lda,
                       const unsigned short* __restrict__ Wt,
                       const float* __restrict__ bias,
                       float* __restrict__ C, int ldc,
                       int M, int N, int K, int relu)
{
    __shared__ __align__(32) unsigned short As[TM * TK];
    __shared__ __align__(32) unsigned short Bs[TN * TK];
    const int tid = threadIdx.x, lane = tid & 31, wave = tid >> 5;
    const int wm = (wave >> 1) * 32, wn = (wave & 1) * 64;
    const int rowBase = blockIdx.y * TM, colBase = blockIdx.x * TN;
    v8f acc[2][4] = {};
    const int kTiles = K / TK;
    for (int kb = 0; kb < kTiles; ++kb) {
        const int k0 = kb * TK;
        stage_A(A, lda, M, K, rowBase, k0, kTiles, kb, As, tid);
#if HAVE_TDM
        if (wave == 0) {
            // Tensor DMA descriptor (ISA ch.8): 2D bf16 tile, tile_dim0=TK cols,
            // tile_dim1=TN rows of Wt, row stride = K elements, packed into LDS.
            const unsigned long long ga =
                (unsigned long long)(size_t)(Wt + (size_t)colBase * K + k0);
            const unsigned lds_off = (unsigned)(size_t)(void*)Bs;
            u32x4 g0;
            g0[0] = 1u;                                          // count=1 (valid)
            g0[1] = lds_off;                                     // lds_addr
            g0[2] = (unsigned)(ga & 0xFFFFFFFFu);                // global_addr[31:0]
            g0[3] = (unsigned)((ga >> 32) & 0x01FFFFFFu)         // global_addr[56:32]
                  | (2u << 30);                                  // type=2 (image)
            const unsigned td0 = (unsigned)K;                    // tensor_dim0
            const unsigned td1 = (unsigned)N;                    // tensor_dim1
            i32x8 g1;
            g1[0] = (int)(1u << 16);                             // data_size=1 (2B)
            g1[1] = (int)((td0 & 0xFFFFu) << 16);                // dim0[15:0]
            g1[2] = (int)(((td0 >> 16) & 0xFFFFu) | ((td1 & 0xFFFFu) << 16));
            g1[3] = (int)(((td1 >> 16) & 0xFFFFu) | ((unsigned)TK << 16)); // tile_dim0
            g1[4] = (int)(unsigned)TN;                           // tile_dim1 (tile_dim2=0)
            g1[5] = (int)(unsigned)K;                            // dim0_stride[31:0]
            g1[6] = 0;                                           // stride[47:32] | dim1_stride lo
            g1[7] = 0;
            const i32x4 z4 = {0, 0, 0, 0};
            const i32x8 z8 = {0, 0, 0, 0, 0, 0, 0, 0};
            __builtin_amdgcn_tensor_load_to_lds(g0, g1, z4, z4, z8, 0);
            __builtin_amdgcn_s_wait_tensorcnt(0);
        }
#else
        {   // fallback: plain bf16 copy of the transposed weight tile
            const int n = tid;                   // 128 threads -> 128 rows
            const unsigned* src = (const unsigned*)(Wt + (size_t)(colBase + n) * K + k0);
            unsigned* dst = (unsigned*)&Bs[n * TK];
            #pragma unroll
            for (int j = 0; j < 16; ++j) dst[j] = src[j];
        }
#endif
        __syncthreads();
        wmma_tile(As, Bs, lane, wm, wn, acc);
        __syncthreads();
    }
    epilogue(acc, bias, C, ldc, M, N, rowBase, colBase, lane, wm, wn, relu);
}

// ---- weight fp32 [K,N] -> bf16 transposed [N,K] ---------------------------
__global__ void convert_wt(const float* __restrict__ W, int K, int N,
                           unsigned short* __restrict__ Wt)
{
    const size_t i = (size_t)blockIdx.x * blockDim.x + threadIdx.x;
    if (i >= (size_t)K * N) return;
    const int n = (int)(i / K), k = (int)(i % K);
    Wt[i] = f2bf(W[(size_t)k * N + n]);
}

// ---------------- BatchNorm (training-mode batch stats) --------------------
__global__ __launch_bounds__(256)
void bn_stats(const float* __restrict__ H, int rows, int cols,
              float* __restrict__ mean, float* __restrict__ istd)
{
    const int c = blockIdx.x;
    float s = 0.f, sq = 0.f;
    for (int r = threadIdx.x; r < rows; r += blockDim.x) {
        const float v = H[(size_t)r * cols + c];
        s += v; sq += v * v;
    }
    __shared__ float sh[256], sh2[256];
    sh[threadIdx.x] = s; sh2[threadIdx.x] = sq;
    __syncthreads();
    for (int st = 128; st > 0; st >>= 1) {
        if (threadIdx.x < st) {
            sh[threadIdx.x]  += sh[threadIdx.x + st];
            sh2[threadIdx.x] += sh2[threadIdx.x + st];
        }
        __syncthreads();
    }
    if (threadIdx.x == 0) {
        const float mu  = sh[0] / (float)rows;
        const float var = sh2[0] / (float)rows - mu * mu;
        mean[c] = mu;
        istd[c] = rsqrtf(var + 1e-5f);
    }
}

__global__ void bn_apply(float* __restrict__ H, int rows, int cols,
                         const float* __restrict__ mean, const float* __restrict__ istd,
                         const float* __restrict__ gamma, const float* __restrict__ beta)
{
    const size_t i = (size_t)blockIdx.x * blockDim.x + threadIdx.x;
    if (i >= (size_t)rows * cols) return;
    const int c = (int)(i % cols);
    H[i] = (H[i] - mean[c]) * istd[c] * gamma[c] + beta[c];
}

// ---------------- FPS: greedy farthest-point walk (reference variant) ------
__global__ __launch_bounds__(256)
void fps_kernel(const float* __restrict__ pos, int N, int m, int* __restrict__ idxOut)
{
    const int b = blockIdx.x;
    const float* P = pos + (size_t)b * N * 3;
    __shared__ float px[1024], py[1024], pz[1024];
    __shared__ unsigned char vis[1024];
    __shared__ float rv[256];
    __shared__ int   ri[256];
    __shared__ int   s_prev;

    for (int j = threadIdx.x; j < N; j += 256) {
        px[j] = P[j * 3 + 0]; py[j] = P[j * 3 + 1]; pz[j] = P[j * 3 + 2];
        vis[j] = 0;
    }
    if (threadIdx.x == 0) s_prev = 0;
    __syncthreads();

    for (int step = 1; step < m; ++step) {
        const int prev = s_prev;
        if (threadIdx.x == 0) vis[prev] = 1;
        __syncthreads();
        const float cx = px[prev], cy = py[prev], cz = pz[prev];
        float best = -3.0e38f; int bidx = 0;
        for (int j = threadIdx.x; j < N; j += 256) {
            const float dx = cx - px[j], dy = cy - py[j], dz = cz - pz[j];
            float d = sqrtf(fmaxf(dx * dx + dy * dy + dz * dz, 1e-12f));
            if (j == prev) d -= 1e8f;
            const float row = vis[j] ? -1e-8f : d;
            if (row > best || (row == best && j < bidx)) { best = row; bidx = j; }
        }
        rv[threadIdx.x] = best; ri[threadIdx.x] = bidx;
        __syncthreads();
        for (int st = 128; st > 0; st >>= 1) {
            if (threadIdx.x < st) {
                const float ov = rv[threadIdx.x + st];
                const int   oi = ri[threadIdx.x + st];
                if (ov > rv[threadIdx.x] ||
                    (ov == rv[threadIdx.x] && oi < ri[threadIdx.x])) {
                    rv[threadIdx.x] = ov; ri[threadIdx.x] = oi;
                }
            }
            __syncthreads();
        }
        if (threadIdx.x == 0) s_prev = ri[0];
        __syncthreads();
    }
    if (threadIdx.x == 0) {
        vis[s_prev] = 1;
        int cnt = 0;
        for (int j = 0; j < N; ++j)
            if (vis[j]) idxOut[(size_t)b * m + (cnt++)] = j;   // sorted ascending
    }
}

// ---------------- kNN: k smallest distances per FPS center -----------------
__global__ __launch_bounds__(256)
void knn_kernel(const float* __restrict__ pos, const int* __restrict__ fpsIdx,
                int N, int m, int k, int* __restrict__ nbr)
{
    const int g = blockIdx.x;
    const int b = g / m;
    const float* P = pos + (size_t)b * N * 3;
    const int ci = fpsIdx[g];
    __shared__ float d[1024];
    __shared__ float rv[256];
    __shared__ int   ri[256];
    const float cx = P[ci * 3 + 0], cy = P[ci * 3 + 1], cz = P[ci * 3 + 2];
    for (int j = threadIdx.x; j < N; j += 256) {
        const float dx = cx - P[j * 3 + 0], dy = cy - P[j * 3 + 1], dz = cz - P[j * 3 + 2];
        float dd = sqrtf(fmaxf(dx * dx + dy * dy + dz * dz, 1e-12f));
        if (j == ci) dd -= 1e8f;            // diagonal mask -> self is nearest
        d[j] = dd;
    }
    __syncthreads();
    for (int kk = 0; kk < k; ++kk) {
        float best = 3.0e38f; int bidx = 0;
        for (int j = threadIdx.x; j < N; j += 256) {
            if (d[j] < best || (d[j] == best && j < bidx)) { best = d[j]; bidx = j; }
        }
        rv[threadIdx.x] = best; ri[threadIdx.x] = bidx;
        __syncthreads();
        for (int st = 128; st > 0; st >>= 1) {
            if (threadIdx.x < st) {
                const float ov = rv[threadIdx.x + st];
                const int   oi = ri[threadIdx.x + st];
                if (ov < rv[threadIdx.x] ||
                    (ov == rv[threadIdx.x] && oi < ri[threadIdx.x])) {
                    rv[threadIdx.x] = ov; ri[threadIdx.x] = oi;
                }
            }
            __syncthreads();
        }
        if (threadIdx.x == 0) { nbr[(size_t)g * k + kk] = ri[0]; d[ri[0]] = 3.0e38f; }
        __syncthreads();
    }
}

// ---------------- gathers / concatenations ---------------------------------
__global__ void gather_pos(const float* __restrict__ pos, const int* __restrict__ fpsIdx,
                           int N, int m, int total, float* __restrict__ pos_c)
{
    const int i = blockIdx.x * blockDim.x + threadIdx.x;
    if (i >= total) return;
    const int dd = i % 3, g = i / 3, b = g / m;
    pos_c[i] = pos[((size_t)b * N + fpsIdx[g]) * 3 + dd];
}

__global__ void build_msg(const float* __restrict__ pos, const float* __restrict__ x, int C,
                          const int* __restrict__ fpsIdx, const int* __restrict__ nbr,
                          int N, int m, int k, int rows, float* __restrict__ msg)
{
    const int cols = 3 + C;
    const size_t i = (size_t)blockIdx.x * blockDim.x + threadIdx.x;
    if (i >= (size_t)rows * cols) return;
    const int row = (int)(i / cols), col = (int)(i % cols);
    const int b  = row / (m * k);
    const int ii = (row / k) % m;
    const int ni = nbr[row];
    if (col < 3) {
        const int ci = fpsIdx[b * m + ii];
        msg[i] = pos[((size_t)b * N + ni) * 3 + col] - pos[((size_t)b * N + ci) * 3 + col];
    } else {
        msg[i] = x[((size_t)b * N + ni) * C + (col - 3)];
    }
}

__global__ __launch_bounds__(256)
void aggr_concat(const float* __restrict__ x, int N, int C,
                 const int* __restrict__ fpsIdx, int m,
                 const float* __restrict__ npts, int Cl,
                 const float* __restrict__ w, int Cw, int k,
                 float* __restrict__ out)
{
    const int g  = blockIdx.x;
    const int b  = g / m;
    const int ci = fpsIdx[g];
    const int ld = C + Cl * Cw;
    const float* xr = x + ((size_t)b * N + ci) * C;
    for (int t = threadIdx.x; t < ld; t += blockDim.x) {
        float v;
        if (t < C) {
            v = xr[t];
        } else {
            const int a = t - C, c = a / Cw, wi = a % Cw;
            const float* np = npts + (size_t)g * k * Cl + c;
            const float* wp = w    + (size_t)g * k * Cw + wi;
            float s = 0.f;
            for (int j = 0; j < k; ++j) s += np[(size_t)j * Cl] * wp[(size_t)j * Cw];
            v = s;
        }
        out[(size_t)g * ld + t] = v;
    }
}

__global__ void concat2(const float* __restrict__ A, int ca,
                        const float* __restrict__ Bm, int cb,
                        int rows, float* __restrict__ out)
{
    const int cols = ca + cb;
    const size_t i = (size_t)blockIdx.x * blockDim.x + threadIdx.x;
    if (i >= (size_t)rows * cols) return;
    const int r = (int)(i / cols), c = (int)(i % cols);
    out[i] = (c < ca) ? A[(size_t)r * ca + c] : Bm[(size_t)r * cb + (c - ca)];
}

__global__ __launch_bounds__(256)
void maxpool_rows(const float* __restrict__ H, int Mrows, int D, float* __restrict__ out)
{
    const int b = blockIdx.x;
    for (int dd = threadIdx.x; dd < D; dd += blockDim.x) {
        float mx = -3.0e38f;
        for (int r = 0; r < Mrows; ++r)
            mx = fmaxf(mx, H[((size_t)b * Mrows + r) * D + dd]);
        out[(size_t)b * D + dd] = mx;
    }
}

__global__ void log_softmax_rows(const float* __restrict__ logits, float* __restrict__ out, int cols)
{
    const int r = blockIdx.x;
    if (threadIdx.x != 0) return;
    float mx = -3.0e38f;
    for (int c = 0; c < cols; ++c) mx = fmaxf(mx, logits[r * cols + c]);
    float s = 0.f;
    for (int c = 0; c < cols; ++c) s += expf(logits[r * cols + c] - mx);
    const float ls = logf(s);
    for (int c = 0; c < cols; ++c) out[r * cols + c] = logits[r * cols + c] - mx - ls;
}

// ---------------------------------------------------------------------------
extern "C" void kernel_launch(void* const* d_in, const int* in_sizes, int n_in,
                              void* d_out, int out_size, void* d_ws, size_t ws_size,
                              hipStream_t stream)
{
    (void)in_sizes; (void)n_in; (void)out_size; (void)ws_size;

    const int B = 16, N1 = 1024, m1 = 512, k1 = 32, C1 = 3;
    const int N2 = 512, m2 = 128, k2 = 64, C2 = 256;
    const int bm1 = B * m1, bm2 = B * m2;
    const int rows1 = bm1 * k1, rows2 = bm2 * k2;

    const float* pos = (const float*)d_in[0];
    const float* x   = (const float*)d_in[1];

    struct LayerP { const float *W, *b, *g, *be; };
    LayerP L[14];   // w1(0,1) l1(2,3) g1(4,5) w2(6,7) l2(8,9) g2(10,11) pool(12,13)
    int p = 2;
    for (int i = 0; i < 14; ++i) {
        L[i].W  = (const float*)d_in[p++];
        L[i].b  = (const float*)d_in[p++];
        L[i].g  = (const float*)d_in[p++];
        L[i].be = (const float*)d_in[p++];
    }
    const float* lin1W = (const float*)d_in[p++]; const float* lin1b = (const float*)d_in[p++];
    const float* lin2W = (const float*)d_in[p++]; const float* lin2b = (const float*)d_in[p++];
    const float* lin3W = (const float*)d_in[p++]; const float* lin3b = (const float*)d_in[p++];

    // ---- workspace bump allocator ----
    char* wsp = (char*)d_ws;
    auto alloc = [&](size_t bytes) -> void* {
        void* r = (void*)wsp;
        wsp += (bytes + 255) & ~(size_t)255;
        return r;
    };
    float* bn_mean = (float*)alloc(2048 * sizeof(float));
    float* bn_istd = (float*)alloc(2048 * sizeof(float));

    int*   fps1  = (int*)  alloc((size_t)bm1 * sizeof(int));
    int*   nbr1  = (int*)  alloc((size_t)rows1 * sizeof(int));
    float* posc1 = (float*)alloc((size_t)bm1 * 3 * sizeof(float));
    float* msg1  = (float*)alloc((size_t)rows1 * 6 * sizeof(float));
    float* wh1   = (float*)alloc((size_t)rows1 * 8 * sizeof(float));
    float* wout1 = (float*)alloc((size_t)rows1 * 4 * sizeof(float));
    float* lh1   = (float*)alloc((size_t)rows1 * 16 * sizeof(float));
    float* lout1 = (float*)alloc((size_t)rows1 * 16 * sizeof(float));
    float* cat1  = (float*)alloc((size_t)bm1 * 67 * sizeof(float));
    float* gh1   = (float*)alloc((size_t)bm1 * 256 * sizeof(float));
    float* h1    = (float*)alloc((size_t)bm1 * 256 * sizeof(float));

    int*   fps2  = (int*)  alloc((size_t)bm2 * sizeof(int));
    int*   nbr2  = (int*)  alloc((size_t)rows2 * sizeof(int));
    float* posc2 = (float*)alloc((size_t)bm2 * 3 * sizeof(float));
    float* msg2  = (float*)alloc((size_t)rows2 * 259 * sizeof(float));  // ~136 MB
    float* wh2   = (float*)alloc((size_t)rows2 * 8 * sizeof(float));
    float* wout2 = (float*)alloc((size_t)rows2 * 4 * sizeof(float));
    float* lh2   = (float*)alloc((size_t)rows2 * 256 * sizeof(float));  // ~134 MB
    float* lout2 = msg2;                        // alias: msg2 dead after l2 GEMM0
    float* cat2  = (float*)alloc((size_t)bm2 * 1280 * sizeof(float));
    float* gh2   = (float*)alloc((size_t)bm2 * 1024 * sizeof(float));
    float* h2    = (float*)alloc((size_t)bm2 * 2048 * sizeof(float));
    float* plin  = (float*)alloc((size_t)bm2 * 2051 * sizeof(float));
    float* ph    = (float*)alloc((size_t)bm2 * 2048 * sizeof(float));
    float* pout  = (float*)alloc((size_t)bm2 * 2048 * sizeof(float));
    float* gmax  = (float*)alloc((size_t)B * 2048 * sizeof(float));
    float* fc1   = (float*)alloc((size_t)B * 512 * sizeof(float));
    float* fc2   = (float*)alloc((size_t)B * 256 * sizeof(float));
    float* lgts  = (float*)alloc((size_t)B * 10 * sizeof(float));

    // ---- pre-convert eligible weights to transposed bf16 (weight-stationary)
    auto conv_wt = [&](const float* W, int K, int N) -> unsigned short* {
        unsigned short* Wt = (unsigned short*)alloc((size_t)K * N * sizeof(unsigned short));
        size_t tot = (size_t)K * N;
        convert_wt<<<(unsigned)((tot + 255) / 256), 256, 0, stream>>>(W, K, N, Wt);
        return Wt;
    };
    unsigned short* Wt_g1b  = conv_wt(L[5].W, 256, 256);
    unsigned short* Wt_l2b  = conv_wt(L[9].W, 256, 256);
    unsigned short* Wt_g2a  = conv_wt(L[10].W, 1280, 1024);
    unsigned short* Wt_g2b  = conv_wt(L[11].W, 1024, 2048);
    unsigned short* Wt_plb  = conv_wt(L[13].W, 2048, 2048);
    unsigned short* Wt_lin1 = conv_wt(lin1W, 2048, 512);
    unsigned short* Wt_lin2 = conv_wt(lin2W, 512, 256);

    // ---- launch helpers ----
    auto run_linear = [&](const float* A, int lda, const float* W, const float* bias,
                          float* C, int M, int N, int K, int relu) {
        dim3 grid((N + TN - 1) / TN, (M + TM - 1) / TM);
        gemm_bias_act<<<grid, 128, 0, stream>>>(A, lda, W, bias, C, N, M, N, K, relu);
    };
    auto run_linear_tdm = [&](const float* A, int lda, const unsigned short* Wt, const float* bias,
                              float* C, int M, int N, int K, int relu) {
        dim3 grid(N / TN, (M + TM - 1) / TM);
        gemm_bias_act_tdm<<<grid, 128, 0, stream>>>(A, lda, Wt, bias, C, N, M, N, K, relu);
    };
    auto run_bn = [&](float* H, int rows, int cols, const float* g, const float* be) {
        bn_stats<<<cols, 256, 0, stream>>>(H, rows, cols, bn_mean, bn_istd);
        size_t tot = (size_t)rows * cols;
        bn_apply<<<(unsigned)((tot + 255) / 256), 256, 0, stream>>>(H, rows, cols, bn_mean, bn_istd, g, be);
    };
    auto run_mlp2 = [&](const float* A, int lda, int M, int K,
                        LayerP& l0, int H0, LayerP& l1, int H1,
                        float* buf0, float* buf1) {
        run_linear(A, lda, l0.W, l0.b, buf0, M, H0, K, 1); run_bn(buf0, M, H0, l0.g, l0.be);
        run_linear(buf0, H0, l1.W, l1.b, buf1, M, H1, H0, 1); run_bn(buf1, M, H1, l1.g, l1.be);
    };

    // =================== LieConv layer 1 ===================
    fps_kernel<<<B, 256, 0, stream>>>(pos, N1, m1, fps1);
    gather_pos<<<(bm1 * 3 + 255) / 256, 256, 0, stream>>>(pos, fps1, N1, m1, bm1 * 3, posc1);
    knn_kernel<<<bm1, 256, 0, stream>>>(pos, fps1, N1, m1, k1, nbr1);
    {
        size_t tot = (size_t)rows1 * (3 + C1);
        build_msg<<<(unsigned)((tot + 255) / 256), 256, 0, stream>>>(pos, x, C1, fps1, nbr1, N1, m1, k1, rows1, msg1);
    }
    run_mlp2(msg1, 6, rows1, 3, L[0], 8, L[1], 4, wh1, wout1);          // weightnet w1
    run_mlp2(msg1, 6, rows1, 6, L[2], 16, L[3], 16, lh1, lout1);        // local_nn l1
    aggr_concat<<<bm1, 256, 0, stream>>>(x, N1, C1, fps1, m1, lout1, 16, wout1, 4, k1, cat1);
    // global_nn g1: layer0 ragged K=67 (generic), layer1 256x256 (TDM)
    run_linear(cat1, 67, L[4].W, L[4].b, gh1, bm1, 256, 67, 1); run_bn(gh1, bm1, 256, L[4].g, L[4].be);
    run_linear_tdm(gh1, 256, Wt_g1b, L[5].b, h1, bm1, 256, 256, 1); run_bn(h1, bm1, 256, L[5].g, L[5].be);

    // =================== LieConv layer 2 ===================
    fps_kernel<<<B, 256, 0, stream>>>(posc1, N2, m2, fps2);
    gather_pos<<<(bm2 * 3 + 255) / 256, 256, 0, stream>>>(posc1, fps2, N2, m2, bm2 * 3, posc2);
    knn_kernel<<<bm2, 256, 0, stream>>>(posc1, fps2, N2, m2, k2, nbr2);
    {
        size_t tot = (size_t)rows2 * (3 + C2);
        build_msg<<<(unsigned)((tot + 255) / 256), 256, 0, stream>>>(posc1, h1, C2, fps2, nbr2, N2, m2, k2, rows2, msg2);
    }
    run_mlp2(msg2, 259, rows2, 3, L[6], 8, L[7], 4, wh2, wout2);        // weightnet w2
    // local_nn l2 (~35 GFLOP): layer0 ragged K=259 (generic), layer1 TDM
    run_linear(msg2, 259, L[8].W, L[8].b, lh2, rows2, 256, 259, 1); run_bn(lh2, rows2, 256, L[8].g, L[8].be);
    run_linear_tdm(lh2, 256, Wt_l2b, L[9].b, lout2, rows2, 256, 256, 1); run_bn(lout2, rows2, 256, L[9].g, L[9].be);
    aggr_concat<<<bm2, 256, 0, stream>>>(h1, N2, C2, fps2, m2, lout2, 256, wout2, 4, k2, cat2);
    // global_nn g2: both layers TDM
    run_linear_tdm(cat2, 1280, Wt_g2a, L[10].b, gh2, bm2, 1024, 1280, 1); run_bn(gh2, bm2, 1024, L[10].g, L[10].be);
    run_linear_tdm(gh2, 1024, Wt_g2b, L[11].b, h2, bm2, 2048, 1024, 1); run_bn(h2, bm2, 2048, L[11].g, L[11].be);

    // =================== pool MLP + global max ===================
    {
        size_t tot = (size_t)bm2 * 2051;
        concat2<<<(unsigned)((tot + 255) / 256), 256, 0, stream>>>(h2, 2048, posc2, 3, bm2, plin);
    }
    run_linear(plin, 2051, L[12].W, L[12].b, ph, bm2, 2048, 2051, 1); run_bn(ph, bm2, 2048, L[12].g, L[12].be);
    run_linear_tdm(ph, 2048, Wt_plb, L[13].b, pout, bm2, 2048, 2048, 1); run_bn(pout, bm2, 2048, L[13].g, L[13].be);
    maxpool_rows<<<B, 256, 0, stream>>>(pout, m2, 2048, gmax);

    // =================== classifier head ===================
    run_linear_tdm(gmax, 2048, Wt_lin1, lin1b, fc1, B, 512, 2048, 1);
    run_linear_tdm(fc1, 512, Wt_lin2, lin2b, fc2, B, 256, 512, 1);
    run_linear(fc2, 256, lin3W, lin3b, lgts, B, 10, 256, 0);
    log_softmax_rows<<<B, 1, 0, stream>>>(lgts, (float*)d_out, 10);
}
